// Encoder_36928128811708
// MI455X (gfx1250) — matrix operand order repaired and checked by reading, not compiled, not run on hardware
//
#include <hip/hip_runtime.h>
#include <math.h>

// ---------------------------------------------------------------------------
// CDNA5 / gfx1250 transformer encoder layer.
// bf16 WMMA for all matmuls; Tensor Data Mover (TDM) for all tile staging.
// ---------------------------------------------------------------------------

typedef __bf16 bf16;
typedef bf16  v16bf __attribute__((ext_vector_type(16)));
typedef bf16  v8bf  __attribute__((ext_vector_type(8)));
typedef float v8f   __attribute__((ext_vector_type(8)));

typedef unsigned int u32x4 __attribute__((ext_vector_type(4)));
typedef int          i32x4 __attribute__((ext_vector_type(4)));
typedef int          i32x8 __attribute__((ext_vector_type(8)));

#define WMMA_BF16(A_, B_, C_) \
    __builtin_amdgcn_wmma_f32_16x16x32_bf16(false, (A_), false, (B_), (short)0, (C_), false, false)

constexpr int D_MODEL = 768, HEADS = 12, HEAD_DIM = 64, HIDDEN = 3072;
constexpr int BATCH = 32, SEQ = 512;
constexpr int MROWS = BATCH * SEQ;          // 16384 token rows
constexpr float LN_EPS = 1e-5f;

// Epilogue modes for the generic GEMM
constexpr int EPI_F32       = 0;  // (+bias) store fp32 row-major
constexpr int EPI_GELU_BF16 = 1;  // +bias, exact gelu, store bf16 row-major
constexpr int EPI_QK        = 2;  // scatter to [B,H,S,64] bf16
constexpr int EPI_V         = 3;  // scatter to [B,H,64,S] bf16 (transposed V)

// ---------------------------------------------------------------------------
// Tensor Data Mover: async 2D bf16 tile load global -> LDS.
// Packs the D# per CDNA5 ISA 8.3/8.4: group0 {count=1, lds_addr, global_addr,
// type=2}; group1 {data_size=1(2B), tensor_dim0/1, tile_dim0/1, dim0 stride}.
// Issued by one wave; completion via TENSORcnt (s_wait_tensorcnt).
// ---------------------------------------------------------------------------
__device__ __forceinline__ void tdm_load_2d(unsigned lds_off, const bf16* gptr,
                                            unsigned tile_d0, unsigned tile_d1,
                                            unsigned tensor_d0, unsigned tensor_d1,
                                            unsigned stride0_elems) {
    unsigned long long ga = (unsigned long long)(uintptr_t)gptr;
    u32x4 g0;
    g0[0] = 1u;                                        // count=1 (valid), user mode
    g0[1] = lds_off;                                   // LDS byte address
    g0[2] = (unsigned)(ga & 0xffffffffu);              // global_addr[31:0]
    g0[3] = (unsigned)((ga >> 32) & 0x01ffffffu)       // global_addr[56:32]
          | (2u << 30);                                // type = 2 ("image")
    i32x8 g1;
    g1[0] = (int)(1u << 16);                           // data_size=1 (2B); wg_mask=0
    g1[1] = (int)((tensor_d0 & 0xffffu) << 16);        // tensor_dim0[15:0] @ bits 63:48
    g1[2] = (int)(((tensor_d0 >> 16) & 0xffffu) |      // tensor_dim0[31:16]
                  ((tensor_d1 & 0xffffu) << 16));      // tensor_dim1[15:0]
    g1[3] = (int)(((tensor_d1 >> 16) & 0xffffu) |      // tensor_dim1[31:16]
                  ((tile_d0 & 0xffffu) << 16));        // tile_dim0 @ bits 127:112
    g1[4] = (int)(tile_d1 & 0xffffu);                  // tile_dim1; tile_dim2=0
    g1[5] = (int)stride0_elems;                        // tensor_dim0_stride[31:0]
    g1[6] = 0;                                         // stride0[47:32]=0; stride1 lo=0
    g1[7] = 0;                                         // stride1 hi = 0
    i32x4 z4 = {0, 0, 0, 0};
#if __clang_major__ >= 23
    i32x8 z8 = {0, 0, 0, 0, 0, 0, 0, 0};
    __builtin_amdgcn_tensor_load_to_lds(g0, g1, z4, z4, z8, 0);
#else
    __builtin_amdgcn_tensor_load_to_lds(g0, g1, z4, z4, 0);
#endif
}

__device__ __forceinline__ unsigned lds_offset_of(const void* p) {
    // Generic (flat) shared-aperture addresses carry the LDS byte offset in [31:0].
    return (unsigned)(uintptr_t)p;
}

// ---------------------------------------------------------------------------
// Elementwise converters
// ---------------------------------------------------------------------------
__global__ void convert_bf16_kernel(const float* __restrict__ in, bf16* __restrict__ out, int n) {
    int i = blockIdx.x * 256 + threadIdx.x;
    if (i < n) out[i] = (bf16)in[i];
}

// dst[b][n][k] = (bf16) src[b][k][n]  -- builds B^T weight layout
__global__ void transpose_bf16_kernel(const float* __restrict__ in, bf16* __restrict__ out,
                                      int batch, int K, int N) {
    int i = blockIdx.x * 256 + threadIdx.x;
    int total = batch * K * N;
    if (i >= total) return;
    int b = i / (K * N);
    int r = i - b * (K * N);
    int n = r / K;
    int k = r - n * K;
    out[i] = (bf16)in[(size_t)b * K * N + (size_t)k * N + n];
}

// ---------------------------------------------------------------------------
// Generic bf16 WMMA GEMM:  C[M,N] = A[M,K] * B[K,N],  B given transposed (Bt[N][K]).
// Block tile 128x128, BK=32, 8 waves, each wave computes 64x32 via 4x2 WMMAs.
// Tiles staged by TDM (wave0 -> A, wave1 -> B), double-buffered so the DMA for
// step k+1 overlaps the WMMAs of step k.
// ---------------------------------------------------------------------------
template <int N, int K, int EPI>
__global__ __launch_bounds__(256) void gemm_wmma_kernel(
    const bf16* __restrict__ A, const bf16* __restrict__ Bt,
    const float* __restrict__ bias, float* __restrict__ Cf, bf16* __restrict__ Cb) {
    __shared__ alignas(32) bf16 As[2][128][32];
    __shared__ alignas(32) bf16 Bs[2][128][32];

    const int tid  = threadIdx.x;
    const int lane = tid & 31;
    const int wave = tid >> 5;
    const int row0 = blockIdx.x * 128;
    const int col0 = blockIdx.y * 128;
    const int wm   = (wave >> 2) * 64;  // wave M offset inside tile (0 / 64)
    const int wn   = (wave & 3) * 32;   // wave N offset inside tile (0..96)

    v8f acc[4][2] = {};

    auto stage = [&](int buf, int kk) {
        if (wave == 0) {
            tdm_load_2d(lds_offset_of(&As[buf][0][0]), A + (size_t)row0 * K + kk,
                        /*tile*/ 32, 128, /*tensor*/ (unsigned)(K - kk), 128, (unsigned)K);
        } else if (wave == 1) {
            tdm_load_2d(lds_offset_of(&Bs[buf][0][0]), Bt + (size_t)col0 * K + kk,
                        /*tile*/ 32, 128, /*tensor*/ (unsigned)(K - kk), 128, (unsigned)K);
        }
    };

    stage(0, 0);
    for (int kk = 0; kk < K; kk += 32) {
        const int p = (kk >> 5) & 1;
        __builtin_amdgcn_s_wait_tensorcnt((short)0);  // issuing waves drain TENSORcnt
        __syncthreads();                              // tile visible to all waves
        if (kk + 32 < K) stage(p ^ 1, kk + 32);       // overlap next DMA with compute

        // A fragments: lane holds row (lane&15); half=lane>>4 selects contiguous K half*16..half*16+15
        v16bf af[4];
#pragma unroll
        for (int ms = 0; ms < 4; ++ms) {
            int r = wm + ms * 16 + (lane & 15);
            int half = lane >> 4;
            v8bf lo = *(const v8bf*)&As[p][r][half * 16];
            v8bf hi = *(const v8bf*)&As[p][r][half * 16 + 8];
            af[ms] = __builtin_shufflevector(lo, hi, 0, 1, 2, 3, 4, 5, 6, 7,
                                             8, 9, 10, 11, 12, 13, 14, 15);
        }
#pragma unroll
        for (int ns = 0; ns < 2; ++ns) {
            // B fragment: lane = output column, contiguous 16 K-values at khalf*16
            int c = wn + ns * 16 + (lane & 15);
            v16bf bfrag = *(const v16bf*)&Bs[p][c][(lane >> 4) * 16];
#pragma unroll
            for (int ms = 0; ms < 4; ++ms)
                acc[ms][ns] = WMMA_BF16(af[ms], bfrag, acc[ms][ns]);
        }
        __syncthreads();  // reads done before this buffer is DMA'd again
    }

    // Epilogue: C layout -> lanes 0-15 rows i, lanes 16-31 rows 8+i.
#pragma unroll
    for (int ms = 0; ms < 4; ++ms) {
#pragma unroll
        for (int ns = 0; ns < 2; ++ns) {
#pragma unroll
            for (int i = 0; i < 8; ++i) {
                int r = row0 + wm + ms * 16 + ((lane < 16) ? i : 8 + i);
                int c = col0 + wn + ns * 16 + (lane & 15);
                float v = acc[ms][ns][i];
                if (EPI == EPI_F32) {
                    if (bias) v += bias[c];
                    Cf[(size_t)r * N + c] = v;
                } else if (EPI == EPI_GELU_BF16) {
                    v += bias[c];
                    v = 0.5f * v * (1.0f + erff(v * 0.70710678118654752f));
                    Cb[(size_t)r * N + c] = (bf16)v;
                } else if (EPI == EPI_QK) {
                    int b = r >> 9, s = r & 511, h = c >> 6, d = c & 63;
                    Cb[(((size_t)(b * HEADS + h) * SEQ) + s) * HEAD_DIM + d] = (bf16)v;
                } else {  // EPI_V: store V transposed [B,H,64,S]
                    int b = r >> 9, s = r & 511, h = c >> 6, d = c & 63;
                    Cb[(((size_t)(b * HEADS + h) * HEAD_DIM) + d) * SEQ + s] = (bf16)v;
                }
            }
        }
    }
}

// ---------------------------------------------------------------------------
// Flash attention. grid = (B*H, S/64), block = 128 threads (4 waves).
// Each wave owns 16 query rows; KV processed in tiles of 32 keys staged by TDM
// (wave0 -> K tile, wave1 -> V^T tile), double-buffered.
// q,k: [BH, S, 64] bf16 ; vT: [BH, 64, S] bf16 ; O: [B, S, 768] bf16.
// ---------------------------------------------------------------------------
__global__ __launch_bounds__(128) void flash_attn_kernel(
    const bf16* __restrict__ q, const bf16* __restrict__ k,
    const bf16* __restrict__ vT, bf16* __restrict__ O) {
    __shared__ alignas(32) bf16 Ks[2][32][64];    // 32 keys x 64 head-dim
    __shared__ alignas(32) bf16 Vs[2][64][32];    // V^T: head-dim x 32 keys
    __shared__ alignas(32) bf16 Ps[4][16][32];    // per-wave P tile for layout swap

    const int bh   = blockIdx.x;
    const int b    = bh / HEADS;
    const int h    = bh - b * HEADS;
    const int tid  = threadIdx.x;
    const int wave = tid >> 5;
    const int lane = tid & 31;
    const int q0   = blockIdx.y * 64 + wave * 16;

    const bf16* qb = q  + ((size_t)bh * SEQ + q0) * HEAD_DIM;
    const bf16* kb = k  + (size_t)bh * SEQ * HEAD_DIM;
    const bf16* vb = vT + (size_t)bh * HEAD_DIM * SEQ;

    // Q A-fragments for the two K=32 steps over head_dim.
    v16bf aq0, aq1;
    {
        int r = lane & 15, half = lane >> 4;
        const bf16* p = qb + (size_t)r * HEAD_DIM;
        v8bf l0 = *(const v8bf*)(p + half * 16);
        v8bf h0 = *(const v8bf*)(p + 8 + half * 16);
        v8bf l1 = *(const v8bf*)(p + 32 + half * 16);
        v8bf h1 = *(const v8bf*)(p + 40 + half * 16);
        aq0 = __builtin_shufflevector(l0, h0, 0, 1, 2, 3, 4, 5, 6, 7, 8, 9, 10, 11, 12, 13, 14, 15);
        aq1 = __builtin_shufflevector(l1, h1, 0, 1, 2, 3, 4, 5, 6, 7, 8, 9, 10, 11, 12, 13, 14, 15);
    }

    v8f o0 = {}, o1 = {}, o2 = {}, o3 = {};
    float mrow[8], lrow[8];
#pragma unroll
    for (int i = 0; i < 8; ++i) { mrow[i] = -1e30f; lrow[i] = 0.0f; }

    auto stage = [&](int buf, int kv) {
        if (wave == 0) {
            // K tile: 32 rows x 64 elems, rows contiguous (stride 64).
            tdm_load_2d(lds_offset_of(&Ks[buf][0][0]), kb + (size_t)kv * HEAD_DIM,
                        /*tile*/ 64, 32, /*tensor*/ 64, 32, 64);
        } else if (wave == 1) {
            // V^T tile: 64 rows x 32 elems, row stride SEQ.
            tdm_load_2d(lds_offset_of(&Vs[buf][0][0]), vb + kv,
                        /*tile*/ 32, 64, /*tensor*/ (unsigned)(SEQ - kv), 64, SEQ);
        }
    };

    stage(0, 0);
    for (int kv = 0; kv < SEQ; kv += 32) {
        const int p = (kv >> 5) & 1;
        __builtin_amdgcn_s_wait_tensorcnt((short)0);
        __syncthreads();
        if (kv + 32 < SEQ) stage(p ^ 1, kv + 32);   // overlap next DMA with compute

        // scores = Q * K^T (two 16-key column groups, two K=32 steps over hd)
        v8f sc[2];
#pragma unroll
        for (int nt = 0; nt < 2; ++nt) {
            int key = nt * 16 + (lane & 15);
            int kh  = (lane >> 4) * 16;
            v16bf b0 = *(const v16bf*)&Ks[p][key][kh];
            v16bf b1 = *(const v16bf*)&Ks[p][key][32 + kh];
            v8f a = {};
            a = WMMA_BF16(aq0, b0, a);
            a = WMMA_BF16(aq1, b1, a);
            sc[nt] = a;
        }

        // Online softmax (row = half-wave group of 16 lanes per C-VGPR).
        float p0s[8], p1s[8];
#pragma unroll
        for (int i = 0; i < 8; ++i) {
            float s0 = sc[0][i] * 0.125f;  // 1/sqrt(64)
            float s1 = sc[1][i] * 0.125f;
            float mx = fmaxf(s0, s1);
#pragma unroll
            for (int off = 1; off < 16; off <<= 1)
                mx = fmaxf(mx, __shfl_xor(mx, off, 32));
            float mn = fmaxf(mrow[i], mx);
            float p0 = __expf(s0 - mn);
            float p1 = __expf(s1 - mn);
            float rs = p0 + p1;
#pragma unroll
            for (int off = 1; off < 16; off <<= 1)
                rs += __shfl_xor(rs, off, 32);
            float corr = __expf(mrow[i] - mn);
            lrow[i] = lrow[i] * corr + rs;
            mrow[i] = mn;
            o0[i] *= corr; o1[i] *= corr; o2[i] *= corr; o3[i] *= corr;
            p0s[i] = p0; p1s[i] = p1;
        }

        // Swap P from C-layout to A-layout via per-wave LDS region.
        {
            int rb = (lane < 16) ? 0 : 8;
            int cc = lane & 15;
#pragma unroll
            for (int i = 0; i < 8; ++i) {
                Ps[wave][rb + i][cc]      = (bf16)p0s[i];
                Ps[wave][rb + i][16 + cc] = (bf16)p1s[i];
            }
        }
        asm volatile("s_wait_dscnt 0x0" ::: "memory");
        __builtin_amdgcn_wave_barrier();

        v16bf ap;
        {
            int r = lane & 15, half = lane >> 4;
            v8bf lo = *(const v8bf*)&Ps[wave][r][half * 16];
            v8bf hi = *(const v8bf*)&Ps[wave][r][half * 16 + 8];
            ap = __builtin_shufflevector(lo, hi, 0, 1, 2, 3, 4, 5, 6, 7,
                                         8, 9, 10, 11, 12, 13, 14, 15);
        }

        // O += P * V  (B fragments from transposed V tile: contiguous over keys)
        {
            int nc = lane & 15;
            int kh = (lane >> 4) * 16;
            v16bf vb0 = *(const v16bf*)&Vs[p][nc][kh];
            v16bf vb1 = *(const v16bf*)&Vs[p][16 + nc][kh];
            v16bf vb2 = *(const v16bf*)&Vs[p][32 + nc][kh];
            v16bf vb3 = *(const v16bf*)&Vs[p][48 + nc][kh];
            o0 = WMMA_BF16(ap, vb0, o0);
            o1 = WMMA_BF16(ap, vb1, o1);
            o2 = WMMA_BF16(ap, vb2, o2);
            o3 = WMMA_BF16(ap, vb3, o3);
        }
        __syncthreads();  // all reads of buffer p done before it is DMA'd again
    }

    // Normalize and store concat-head output [B, S, 768] as bf16.
    {
        int rb = (lane < 16) ? 0 : 8;
        int nc = lane & 15;
#pragma unroll
        for (int i = 0; i < 8; ++i) {
            int s = q0 + rb + i;
            float inv = 1.0f / lrow[i];
            size_t base = ((size_t)b * SEQ + s) * D_MODEL + h * HEAD_DIM;
            O[base + 0  + nc] = (bf16)(o0[i] * inv);
            O[base + 16 + nc] = (bf16)(o1[i] * inv);
            O[base + 32 + nc] = (bf16)(o2[i] * inv);
            O[base + 48 + nc] = (bf16)(o3[i] * inv);
        }
    }
}

// ---------------------------------------------------------------------------
// out = residual + LayerNorm(v)*w + b ; optional bf16 mirror for the next GEMM.
// One 256-thread block per 768-wide row.
// ---------------------------------------------------------------------------
__global__ __launch_bounds__(256) void ln_residual_kernel(
    const float* __restrict__ res, const float* __restrict__ v,
    const float* __restrict__ w, const float* __restrict__ bb,
    float* __restrict__ outF, bf16* __restrict__ outB) {
    __shared__ float red[256];
    const int t = threadIdx.x;
    const size_t row = blockIdx.x;
    const float* vr = v + row * D_MODEL;

    float x0 = vr[t], x1 = vr[t + 256], x2 = vr[t + 512];
    red[t] = x0 + x1 + x2;
    __syncthreads();
#pragma unroll
    for (int st = 128; st > 0; st >>= 1) {
        if (t < st) red[t] += red[t + st];
        __syncthreads();
    }
    float mu = red[0] * (1.0f / 768.0f);
    __syncthreads();

    float d0 = x0 - mu, d1 = x1 - mu, d2 = x2 - mu;
    red[t] = d0 * d0 + d1 * d1 + d2 * d2;
    __syncthreads();
#pragma unroll
    for (int st = 128; st > 0; st >>= 1) {
        if (t < st) red[t] += red[t + st];
        __syncthreads();
    }
    float rstd = rsqrtf(red[0] * (1.0f / 768.0f) + LN_EPS);

    const float* rr = res + row * D_MODEL;
    float o0 = rr[t]       + d0 * rstd * w[t]       + bb[t];
    float o1 = rr[t + 256] + d1 * rstd * w[t + 256] + bb[t + 256];
    float o2 = rr[t + 512] + d2 * rstd * w[t + 512] + bb[t + 512];
    outF[row * D_MODEL + t]       = o0;
    outF[row * D_MODEL + t + 256] = o1;
    outF[row * D_MODEL + t + 512] = o2;
    if (outB) {
        outB[row * D_MODEL + t]       = (bf16)o0;
        outB[row * D_MODEL + t + 256] = (bf16)o1;
        outB[row * D_MODEL + t + 512] = (bf16)o2;
    }
}

// ---------------------------------------------------------------------------
// Host-side orchestration (graph-capture safe: only async launches on `stream`).
// ---------------------------------------------------------------------------
extern "C" void kernel_launch(void* const* d_in, const int* in_sizes, int n_in,
                              void* d_out, int out_size, void* d_ws, size_t ws_size,
                              hipStream_t stream) {
    (void)in_sizes; (void)n_in; (void)out_size; (void)ws_size;

    const float* x    = (const float*)d_in[0];
    const float* Wq   = (const float*)d_in[1];
    const float* Wk   = (const float*)d_in[2];
    const float* Wv   = (const float*)d_in[3];
    const float* Wo   = (const float*)d_in[4];
    const float* ln1w = (const float*)d_in[5];
    const float* ln1b = (const float*)d_in[6];
    const float* W1   = (const float*)d_in[7];
    const float* b1   = (const float*)d_in[8];
    const float* W2   = (const float*)d_in[9];
    const float* b2   = (const float*)d_in[10];
    const float* ln2w = (const float*)d_in[11];
    const float* ln2b = (const float*)d_in[12];
    float* out = (float*)d_out;

    // Deterministic workspace carving (256B aligned).
    char* cur = (char*)d_ws;
    auto take = [&](size_t bytes) -> void* {
        void* r = (void*)cur;
        cur += (bytes + 255) & ~(size_t)255;
        return r;
    };
    const size_t tokE = (size_t)MROWS * D_MODEL;     // 12.58M elems
    bf16* xb    = (bf16*)take(tokE * 2);
    bf16* WqT   = (bf16*)take((size_t)D_MODEL * D_MODEL * 2);
    bf16* WkT   = (bf16*)take((size_t)D_MODEL * D_MODEL * 2);
    bf16* WvT   = (bf16*)take((size_t)D_MODEL * D_MODEL * 2);
    bf16* WoT   = (bf16*)take((size_t)D_MODEL * D_MODEL * 2);
    bf16* W1T   = (bf16*)take((size_t)D_MODEL * HIDDEN * 2);
    bf16* W2T   = (bf16*)take((size_t)HIDDEN * D_MODEL * 2);
    bf16* qbuf  = (bf16*)take(tokE * 2);             // [BH,S,64]
    bf16* kbuf  = (bf16*)take(tokE * 2);             // [BH,S,64]
    bf16* vTbuf = (bf16*)take(tokE * 2);             // [BH,64,S]
    bf16* Obuf  = (bf16*)take(tokE * 2);             // [B,S,768]
    float* mha  = (float*)take(tokE * 4);            // reused for FFN output
    float* out1 = (float*)take(tokE * 4);
    bf16* out1b = (bf16*)take(tokE * 2);
    bf16* hbuf  = (bf16*)take((size_t)MROWS * HIDDEN * 2);

    // 1) Precision conversion + weight transposes (tiny, bandwidth-bound).
    convert_bf16_kernel<<<(int)((tokE + 255) / 256), 256, 0, stream>>>(x, xb, (int)tokE);
    transpose_bf16_kernel<<<(HEADS * D_MODEL * HEAD_DIM + 255) / 256, 256, 0, stream>>>(
        Wq, WqT, HEADS, D_MODEL, HEAD_DIM);
    transpose_bf16_kernel<<<(HEADS * D_MODEL * HEAD_DIM + 255) / 256, 256, 0, stream>>>(
        Wk, WkT, HEADS, D_MODEL, HEAD_DIM);
    transpose_bf16_kernel<<<(HEADS * D_MODEL * HEAD_DIM + 255) / 256, 256, 0, stream>>>(
        Wv, WvT, HEADS, D_MODEL, HEAD_DIM);
    transpose_bf16_kernel<<<(D_MODEL * D_MODEL + 255) / 256, 256, 0, stream>>>(
        Wo, WoT, 1, D_MODEL, D_MODEL);
    transpose_bf16_kernel<<<(D_MODEL * HIDDEN + 255) / 256, 256, 0, stream>>>(
        W1, W1T, 1, D_MODEL, HIDDEN);
    transpose_bf16_kernel<<<(HIDDEN * D_MODEL + 255) / 256, 256, 0, stream>>>(
        W2, W2T, 1, HIDDEN, D_MODEL);

    // 2) QKV projections (WMMA + TDM), scatter to attention layouts.
    dim3 g768(MROWS / 128, D_MODEL / 128);
    gemm_wmma_kernel<D_MODEL, D_MODEL, EPI_QK><<<g768, 256, 0, stream>>>(
        xb, WqT, nullptr, nullptr, qbuf);
    gemm_wmma_kernel<D_MODEL, D_MODEL, EPI_QK><<<g768, 256, 0, stream>>>(
        xb, WkT, nullptr, nullptr, kbuf);
    gemm_wmma_kernel<D_MODEL, D_MODEL, EPI_V><<<g768, 256, 0, stream>>>(
        xb, WvT, nullptr, nullptr, vTbuf);

    // 3) Flash attention (WMMA QK^T + PV, online softmax; no SxS tensor).
    flash_attn_kernel<<<dim3(BATCH * HEADS, SEQ / 64), 128, 0, stream>>>(
        qbuf, kbuf, vTbuf, Obuf);

    // 4) Output projection, then LN1 + residual.
    gemm_wmma_kernel<D_MODEL, D_MODEL, EPI_F32><<<g768, 256, 0, stream>>>(
        Obuf, WoT, nullptr, mha, nullptr);
    ln_residual_kernel<<<MROWS, 256, 0, stream>>>(x, mha, ln1w, ln1b, out1, out1b);

    // 5) FFN: GEMM + bias + exact GELU (bf16 h), GEMM + bias, LN2 + residual.
    gemm_wmma_kernel<HIDDEN, D_MODEL, EPI_GELU_BF16>
        <<<dim3(MROWS / 128, HIDDEN / 128), 256, 0, stream>>>(out1b, W1T, b1, nullptr, hbuf);
    gemm_wmma_kernel<D_MODEL, HIDDEN, EPI_F32><<<g768, 256, 0, stream>>>(
        hbuf, W2T, b2, mha, nullptr);
    ln_residual_kernel<<<MROWS, 256, 0, stream>>>(out1, mha, ln2w, ln2b, out, nullptr);
}
